// CausalAttention_32607391711783
// MI455X (gfx1250) — matrix operand order repaired and checked by reading, not compiled, not run on hardware
//
#include <hip/hip_runtime.h>
#include <hip/hip_bf16.h>

typedef __attribute__((ext_vector_type(16))) __bf16 v16bf;
typedef __attribute__((ext_vector_type(8)))  __bf16 v8bf;
typedef __attribute__((ext_vector_type(8)))  float  v8f;

#define NHEAD 16
#define HDIM  64
#define LSEQ  2048
#define BATCH 2
#define CDIM  1024

__device__ __forceinline__ unsigned short f2bf(float f) {
    union { float f; unsigned u; } u; u.f = f;
    unsigned r = u.u + 0x7FFFu + ((u.u >> 16) & 1u);   // round-to-nearest-even
    return (unsigned short)(r >> 16);
}

__device__ __forceinline__ v8f wmma_bf16(v16bf a, v16bf b, v8f c) {
    return __builtin_amdgcn_wmma_f32_16x16x32_bf16(false, a, false, b, (short)0, c,
                                                   false, false);
}

// A-frag (16x32 bf16, row-major source, stride ld elems). Per ISA layout:
// lanes 0-15: M=lane, K runs 0-7 and 16-23 ; lanes 16-31: M=lane-16, K runs 8-15 and 24-31.
__device__ __forceinline__ v16bf load_a16(const unsigned short* base, int ld) {
    int lane = threadIdx.x & 31;
    const unsigned short* p = base + (size_t)(lane & 15) * ld + ((lane >> 4) << 3);
    v8bf lo = *(const v8bf*)p;
    v8bf hi = *(const v8bf*)(p + 16);
    v16bf a;
#pragma unroll
    for (int i = 0; i < 8; ++i) { a[i] = lo[i]; a[i + 8] = hi[i]; }
    return a;
}

// B-frag (32x16 bf16), source stored N-major: row n holds contiguous K. Lane n=lane&15,
// K runs (lane>>4)*8 .. +7 and 16+(lane>>4)*8 .. +7 (same as A-frag) -> two 16B loads.
__device__ __forceinline__ v16bf load_b16(const unsigned short* base, int ld) {
    int lane = threadIdx.x & 31;
    const unsigned short* p = base + (size_t)(lane & 15) * ld + ((lane >> 4) << 3);
    v8bf lo = *(const v8bf*)p;
    v8bf hi = *(const v8bf*)(p + 16);
    v16bf b;
#pragma unroll
    for (int i = 0; i < 8; ++i) { b[i] = lo[i]; b[i + 8] = hi[i]; }
    return b;
}

// ---------------- precision-conversion kernels ----------------
__global__ void cast_f32_bf16(const float* __restrict__ in,
                              unsigned short* __restrict__ out, int n) {
    int t = blockIdx.x * blockDim.x + threadIdx.x;
    if (t < n) out[t] = f2bf(in[t]);
}

// out[n][k] = in[k][n]   (in: rows_in x cols_in row-major; out: cols_in x rows_in)
__global__ void transpose_cast(const float* __restrict__ in,
                               unsigned short* __restrict__ out,
                               int rows_in, int cols_in) {
    int t = blockIdx.x * blockDim.x + threadIdx.x;
    int total = rows_in * cols_in;
    if (t >= total) return;
    int n = t / rows_in;
    int k = t - n * rows_in;
    out[t] = f2bf(in[(size_t)k * cols_in + n]);
}

// ---------------- fused QKV GEMM + RoPE (32x64 wave tile) ----------------
// X: [4096][1024] bf16, Wt: [3072][1024] bf16 (N-major).
// Q,K -> [B,H,L,D] bf16 (RoPE applied, Q pre-scaled by 1/8), V -> [B,H,D,L] bf16.
__global__ void qkv_rope_gemm(const unsigned short* __restrict__ X,
                              const unsigned short* __restrict__ Wt,
                              unsigned short* __restrict__ Q,
                              unsigned short* __restrict__ K,
                              unsigned short* __restrict__ Vt) {
    int wid   = blockIdx.x * (blockDim.x >> 5) + (threadIdx.x >> 5);
    int tileM = wid / 48, tileN = wid % 48;          // 128 x 48 wave tiles
    int m0 = tileM * 32, n0 = tileN * 64;
    int lane = threadIdx.x & 31, colq = lane & 15, half = lane >> 4;

    v8f acc[2][4] = {};
    for (int kk = 0; kk < CDIM; kk += 32) {
        v16bf a0 = load_a16(X + (size_t)m0 * CDIM + kk,        CDIM);
        v16bf a1 = load_a16(X + (size_t)(m0 + 16) * CDIM + kk, CDIM);
        __builtin_prefetch(X + (size_t)m0 * CDIM + kk + 64, 0, 0);
#pragma unroll
        for (int s = 0; s < 4; ++s) {
            v16bf b = load_b16(Wt + (size_t)(n0 + s * 16) * CDIM + kk, CDIM);
            acc[0][s] = wmma_bf16(a0, b, acc[0][s]);
            acc[1][s] = wmma_bf16(a1, b, acc[1][s]);
        }
    }

    int region = n0 / CDIM;              // 0=q 1=k 2=v (64 | 1024, so uniform per tile)
    int h      = (n0 % CDIM) / HDIM;     // whole tile inside one head

#pragma unroll
    for (int mr = 0; mr < 2; ++mr)
#pragma unroll
    for (int s = 0; s < 4; ++s) {
        int dd = s * 16 + colq;
#pragma unroll
        for (int r = 0; r < 8; ++r) {
            int grow = m0 + mr * 16 + r + 8 * half;
            int b    = grow >> 11;          // /2048
            int pos  = grow & (LSEQ - 1);
            float v  = acc[mr][s][r];
            if (region < 2) {
                float partner = __shfl_xor(v, 1, 32);
                float e     = (float)(dd & ~1);
                float theta = (float)pos * __powf(10000.0f, -e * (1.0f / 64.0f));
                float sn = __sinf(theta), cs = __cosf(theta);
                float rv = (dd & 1) ? (partner * sn + v * cs)
                                    : (v * cs - partner * sn);
                size_t idx = (((size_t)(b * NHEAD + h) * LSEQ) + pos) * HDIM + dd;
                if (region == 0) Q[idx] = f2bf(rv * 0.125f);   // fold 1/sqrt(64)
                else             K[idx] = f2bf(rv);
            } else {
                size_t idx = (((size_t)(b * NHEAD + h) * HDIM) + dd) * LSEQ + pos;
                Vt[idx] = f2bf(v);
            }
        }
    }
}

// ---------------- flash attention (one wave = 16 query rows; 64-key blocks) --------------
__global__ void flash_attn(const unsigned short* __restrict__ Q,
                           const unsigned short* __restrict__ K,
                           const unsigned short* __restrict__ Vt,
                           unsigned short* __restrict__ O) {
    __shared__ __align__(128) unsigned short pbuf[8][16][64];   // per-wave P staging

    int wid   = blockIdx.x * (blockDim.x >> 5) + (threadIdx.x >> 5);
    int qtile = wid & 127;
    int h     = (wid >> 7) & (NHEAD - 1);
    int b     = wid >> 11;
    int q0    = qtile * 16;
    int wv    = (threadIdx.x >> 5) & 7;
    int lane = threadIdx.x & 31, colq = lane & 15, half = lane >> 4;

    const unsigned short* Qh = Q  + ((size_t)(b * NHEAD + h) * LSEQ) * HDIM;
    const unsigned short* Kh = K  + ((size_t)(b * NHEAD + h) * LSEQ) * HDIM;
    const unsigned short* Vh = Vt + ((size_t)(b * NHEAD + h) * HDIM) * LSEQ;
    unsigned short* pb = &pbuf[wv][0][0];

    v16bf qa0 = load_a16(Qh + (size_t)q0 * HDIM,      HDIM);   // d 0..31
    v16bf qa1 = load_a16(Qh + (size_t)q0 * HDIM + 32, HDIM);   // d 32..63

    v8f o[4] = {};
    float m[8], l[8];
#pragma unroll
    for (int r = 0; r < 8; ++r) { m[r] = -1e30f; l[r] = 0.0f; }

    for (int j = 0; j < q0 + 16; j += 64) {
        // --- S = Q K^T for 64 keys (four 16-wide subtiles) ---
        v8f s[4];
#pragma unroll
        for (int t = 0; t < 4; ++t) {
            v8f z = {};
            v16bf b0 = load_b16(Kh + (size_t)(j + 16 * t) * HDIM,      HDIM);
            v16bf b1 = load_b16(Kh + (size_t)(j + 16 * t) * HDIM + 32, HDIM);
            z = wmma_bf16(qa0, b0, z);
            z = wmma_bf16(qa1, b1, z);
            s[t] = z;
        }
        // --- online softmax update (amortized over 64 keys) ---
        float corr[8];
#pragma unroll
        for (int r = 0; r < 8; ++r) {
            int qpos = q0 + r + 8 * half;
            float a[4];
#pragma unroll
            for (int t = 0; t < 4; ++t) {
                int kp = j + 16 * t + colq;
                a[t] = (kp <= qpos) ? s[t][r] : -1e30f;
            }
            float mx = fmaxf(fmaxf(a[0], a[1]), fmaxf(a[2], a[3]));
            mx = fmaxf(mx, __shfl_xor(mx, 1, 32));
            mx = fmaxf(mx, __shfl_xor(mx, 2, 32));
            mx = fmaxf(mx, __shfl_xor(mx, 4, 32));
            mx = fmaxf(mx, __shfl_xor(mx, 8, 32));
            float mn = fmaxf(m[r], mx);
            float rs = 0.0f;
            int prow = r + 8 * half;
#pragma unroll
            for (int t = 0; t < 4; ++t) {
                int kp = j + 16 * t + colq;
                float p = (kp <= qpos) ? __expf(a[t] - mn) : 0.0f;
                rs += p;
                pb[prow * 64 + 16 * t + colq] = f2bf(p);   // C-layout -> LDS row-major
            }
            rs += __shfl_xor(rs, 1, 32);
            rs += __shfl_xor(rs, 2, 32);
            rs += __shfl_xor(rs, 4, 32);
            rs += __shfl_xor(rs, 8, 32);
            corr[r] = __expf(m[r] - mn);
            l[r] = l[r] * corr[r] + rs;
            m[r] = mn;
        }
#pragma unroll
        for (int dt = 0; dt < 4; ++dt)
#pragma unroll
            for (int r = 0; r < 8; ++r) o[dt][r] *= corr[r];

        // --- O += P V  (P re-read from LDS in A-frag layout; K-dim = 64 keys) ---
        v16bf pa0 = load_a16(pb,      64);   // keys j..j+31
        v16bf pa1 = load_a16(pb + 32, 64);   // keys j+32..j+63
#pragma unroll
        for (int dt = 0; dt < 4; ++dt) {
            v16bf vb0 = load_b16(Vh + (size_t)(dt * 16) * LSEQ + j,      LSEQ);
            v16bf vb1 = load_b16(Vh + (size_t)(dt * 16) * LSEQ + j + 32, LSEQ);
            o[dt] = wmma_bf16(pa0, vb0, o[dt]);
            o[dt] = wmma_bf16(pa1, vb1, o[dt]);
        }
    }

    float inv[8];
#pragma unroll
    for (int r = 0; r < 8; ++r) inv[r] = 1.0f / l[r];
#pragma unroll
    for (int dt = 0; dt < 4; ++dt)
#pragma unroll
        for (int r = 0; r < 8; ++r) {
            int pos = q0 + r + 8 * half;
            int dd  = dt * 16 + colq;
            O[((size_t)b * LSEQ + pos) * CDIM + h * HDIM + dd] = f2bf(o[dt][r] * inv[r]);
        }
}

// ---------------- output projection: out = A(bf16) @ WprojT(bf16) -> fp32 ----------------
__global__ void proj_gemm(const unsigned short* __restrict__ A,
                          const unsigned short* __restrict__ Wt,
                          float* __restrict__ out) {
    int wid   = blockIdx.x * (blockDim.x >> 5) + (threadIdx.x >> 5);
    int tileM = wid >> 4, tileN = wid & 15;          // 128 x 16 wave tiles
    int m0 = tileM * 32, n0 = tileN * 64;
    int lane = threadIdx.x & 31, colq = lane & 15, half = lane >> 4;

    v8f acc[2][4] = {};
    for (int kk = 0; kk < CDIM; kk += 32) {
        v16bf a0 = load_a16(A + (size_t)m0 * CDIM + kk,        CDIM);
        v16bf a1 = load_a16(A + (size_t)(m0 + 16) * CDIM + kk, CDIM);
        __builtin_prefetch(A + (size_t)m0 * CDIM + kk + 64, 0, 0);
#pragma unroll
        for (int s = 0; s < 4; ++s) {
            v16bf b = load_b16(Wt + (size_t)(n0 + s * 16) * CDIM + kk, CDIM);
            acc[0][s] = wmma_bf16(a0, b, acc[0][s]);
            acc[1][s] = wmma_bf16(a1, b, acc[1][s]);
        }
    }
#pragma unroll
    for (int mr = 0; mr < 2; ++mr)
#pragma unroll
    for (int s = 0; s < 4; ++s)
#pragma unroll
        for (int r = 0; r < 8; ++r) {
            int row = m0 + mr * 16 + r + 8 * half;
            int col = n0 + s * 16 + colq;
            out[(size_t)row * CDIM + col] = acc[mr][s][r];
        }
}

extern "C" void kernel_launch(void* const* d_in, const int* in_sizes, int n_in,
                              void* d_out, int out_size, void* d_ws, size_t ws_size,
                              hipStream_t stream) {
    (void)in_sizes; (void)n_in; (void)out_size; (void)ws_size;
    const float* x     = (const float*)d_in[0];   // [2,2048,1024]
    const float* Wqkv  = (const float*)d_in[1];   // [1024,3072]
    const float* Wproj = (const float*)d_in[2];   // [1024,1024]
    float* out = (float*)d_out;

    char* ws = (char*)d_ws;
    const size_t MB = 1u << 20;
    unsigned short* Xbf    = (unsigned short*)(ws);             //  8 MiB
    unsigned short* WqkvT  = (unsigned short*)(ws +  8 * MB);   //  6 MiB
    unsigned short* WprojT = (unsigned short*)(ws + 14 * MB);   //  2 MiB
    unsigned short* Qb     = (unsigned short*)(ws + 16 * MB);   //  8 MiB
    unsigned short* Kb     = (unsigned short*)(ws + 24 * MB);   //  8 MiB
    unsigned short* Vtb    = (unsigned short*)(ws + 32 * MB);   //  8 MiB
    unsigned short* Ob     = (unsigned short*)(ws + 40 * MB);   //  8 MiB

    const int nX = BATCH * LSEQ * CDIM;                         // 4194304
    cast_f32_bf16<<<nX / 256, 256, 0, stream>>>(x, Xbf, nX);
    transpose_cast<<<(CDIM * 3 * CDIM) / 256, 256, 0, stream>>>(Wqkv, WqkvT, CDIM, 3 * CDIM);
    transpose_cast<<<(CDIM * CDIM) / 256, 256, 0, stream>>>(Wproj, WprojT, CDIM, CDIM);

    // 128 M-tiles x 48 N-tiles = 6144 waves -> 768 blocks of 8 waves
    qkv_rope_gemm<<<768, 256, 0, stream>>>(Xbf, WqkvT, Qb, Kb, Vtb);

    // 2*16*128 = 4096 waves -> 512 blocks
    flash_attn<<<512, 256, 0, stream>>>(Qb, Kb, Vtb, Ob);

    // 128 x 16 = 2048 waves -> 256 blocks
    proj_gemm<<<256, 256, 0, stream>>>(Ob, WprojT, out);
}
